// Network_77421080478507
// MI455X (gfx1250) — matrix operand order repaired and checked by reading, not compiled, not run on hardware
//
#include <hip/hip_runtime.h>
#include <hip/hip_bf16.h>

typedef __attribute__((ext_vector_type(16))) __bf16 v16bf;
typedef __attribute__((ext_vector_type(8)))  __bf16 v8bf;
typedef __attribute__((ext_vector_type(8)))  float  v8f;

#define T_LEN 256
#define BATCH 64
#define U_DIM 512
#define TB    (T_LEN * BATCH)

// ---------------------------------------------------------------------------
// WMMA fragment loaders (wave32, 16x16x32 bf16).
// A-matrix 16x32: lane m=lane&15; lanes<16: elems 0..7 = K kb..kb+7,
// elems 8..15 = K kb+16..kb+23, kb=0; lanes>=16: kb=8.
// B-matrix 32x16: lane n=lane&15; same K layout as A (elems 0..7 = K kb..kb+7,
// elems 8..15 = K kb+16..kb+23, kb=0 for lanes<16, kb=8 for lanes>=16).
// ---------------------------------------------------------------------------
static __device__ __forceinline__ v16bf load_frag_a(const __bf16* base) {
  v8bf lo = *(const v8bf*)(base);
  v8bf hi = *(const v8bf*)(base + 16);
  v16bf r;
#pragma unroll
  for (int i = 0; i < 8; ++i) { r[i] = lo[i]; r[8 + i] = hi[i]; }
  return r;
}

static __device__ __forceinline__ v16bf load_frag_b(const __bf16* base) {
  v8bf lo = *(const v8bf*)(base);
  v8bf hi = *(const v8bf*)(base + 16);
  v16bf r;
#pragma unroll
  for (int i = 0; i < 8; ++i) { r[i] = lo[i]; r[8 + i] = hi[i]; }
  return r;
}

static __device__ __forceinline__ float sigmoidf_(float x) {
  return 1.0f / (1.0f + __expf(-x));
}

// ---------------------------------------------------------------------------
// Embedding gather + concat -> sent bf16 [T*B][320]
// ---------------------------------------------------------------------------
__global__ void embed_kernel(const float* __restrict__ wemb,
                             const float* __restrict__ temb,
                             const int* __restrict__ winds,
                             const int* __restrict__ tinds,
                             __bf16* __restrict__ sent, int total) {
  int idx = blockIdx.x * blockDim.x + threadIdx.x;
  if (idx >= total) return;
  int d  = idx % 320;
  int tb = idx / 320;
  float v = (d < 256) ? wemb[(size_t)winds[tb] * 256 + d]
                      : temb[(size_t)tinds[tb] * 64 + (d - 256)];
  sent[idx] = (__bf16)v;
}

// ---------------------------------------------------------------------------
// LSTM weight prep: W f32 (4, din+512, 512) -> WxT bf16 [2048][din],
//                                              WhT bf16 [2048][512]  (B^T)
// ---------------------------------------------------------------------------
__global__ void prep_lstm_w_kernel(const float* __restrict__ W, int din,
                                   __bf16* __restrict__ WxT,
                                   __bf16* __restrict__ WhT, int total) {
  int idx = blockIdx.x * blockDim.x + threadIdx.x;
  if (idx >= total) return;
  int cols = din + U_DIM;
  int row  = idx / cols;           // k*512 + u
  int col  = idx % cols;           // d
  int k  = row >> 9;
  int uu = row & 511;
  float v = W[((size_t)k * cols + col) * U_DIM + uu];
  if (col < din) WxT[(size_t)row * din + col] = (__bf16)v;
  else           WhT[(size_t)row * U_DIM + (col - din)] = (__bf16)v;
}

// W f32 [K][N] -> WT bf16 [N][K]
__global__ void prep_bt_kernel(const float* __restrict__ W, int K, int N,
                               __bf16* __restrict__ WT, int total) {
  int idx = blockIdx.x * blockDim.x + threadIdx.x;
  if (idx >= total) return;
  int n  = idx % N;
  int kk = idx / N;
  WT[(size_t)n * K + kk] = (__bf16)W[idx];
}

// ---------------------------------------------------------------------------
// Pre-gate GEMM (parallel part of the LSTM): PG = X @ WxT^T + bias, stored in
// recurrence layout pg[t][n][b] bf16 (n = gate*512+u). One wave per 16x16
// tile; grid = (2048/16, 16384/16).
// ---------------------------------------------------------------------------
__global__ __launch_bounds__(32) void pregate_gemm_kernel(
    const __bf16* __restrict__ A, int lda,
    const __bf16* __restrict__ BT,
    const float* __restrict__ bias,
    __bf16* __restrict__ PG, int kchunks) {
  const int ntile = blockIdx.x;
  const int mtile = blockIdx.y;
  const int lane  = threadIdx.x & 31;
  const int lhalf = lane & 15;
  const int kbA   = (lane < 16) ? 0 : 8;
  const int kbB   = (lane < 16) ? 0 : 8;
  const int hrow  = (lane < 16) ? 0 : 8;

  v8f acc;
#pragma unroll
  for (int i = 0; i < 8; ++i) acc[i] = 0.0f;

  const __bf16* arow = A  + (size_t)(mtile * 16 + lhalf) * lda;
  const __bf16* brow = BT + (size_t)(ntile * 16 + lhalf) * lda;

  for (int kc = 0; kc < kchunks; ++kc) {
    v16bf a = load_frag_a(arow + kc * 32 + kbA);
    v16bf b = load_frag_b(brow + kc * 32 + kbB);
    acc = __builtin_amdgcn_wmma_f32_16x16x32_bf16(
        false, a, false, b, (short)0, acc, false, false);
  }

  const int n = ntile * 16 + lhalf;
  const float bn = bias[n];
#pragma unroll
  for (int r = 0; r < 8; ++r) {
    int m = mtile * 16 + hrow + r;
    int t = m >> 6, b = m & 63;
    PG[((size_t)t * 2048 + n) * 64 + b] = (__bf16)(acc[r] + bn);
  }
}

// ---------------------------------------------------------------------------
// One gate pass of the recurrence: acc initialized from pre-gates (which
// already include bias), then acc += h_{t-1} @ Wh[gate]^T via WMMA.
// ---------------------------------------------------------------------------
static __device__ __forceinline__ void gate_gemm(
    int k, const __bf16* __restrict__ pgt, const __bf16* __restrict__ WhT,
    const __bf16* hlds, int u, int lhalf, int kbA, int kbB, int hrow,
    v8f acc[4]) {
#pragma unroll
  for (int mt = 0; mt < 4; ++mt) {
    v8bf p = *(const v8bf*)(pgt + (size_t)(k * U_DIM + u) * 64 + mt * 16 + hrow);
#pragma unroll
    for (int i = 0; i < 8; ++i) acc[mt][i] = (float)p[i];
  }
  for (int kc = 0; kc < (U_DIM >> 5); ++kc) {
    v16bf bfr = load_frag_b(WhT + (size_t)(k * U_DIM + u) * U_DIM + kc * 32 + kbB);
#pragma unroll
    for (int mt = 0; mt < 4; ++mt) {
      v16bf a = load_frag_a(hlds + (mt * 16 + lhalf) * U_DIM + kc * 32 + kbA);
      acc[mt] = __builtin_amdgcn_wmma_f32_16x16x32_bf16(
          false, a, false, bfr, (short)0, acc[mt], false, false);
    }
  }
}

// ---------------------------------------------------------------------------
// Sequential LSTM recurrence. grid.x = 2 (dir 0 fwd, 1 bwd), block = 1024
// (32 waves). Wave w owns u-columns [16w,16w+16) for all 4 gates; each lane
// keeps 32 cell values in registers across all 256 steps; h (64x512 bf16)
// lives in LDS. 4 passes per step (g,i,f,o) keep live VGPRs ~140 (no spill).
// ---------------------------------------------------------------------------
__global__ __launch_bounds__(1024) void lstm_rec_kernel(
    const __bf16* __restrict__ pg0, const __bf16* __restrict__ pg1,
    const __bf16* WhT0, const __bf16* WhT1,
    const float* c00, const float* c01,
    __bf16* out0, __bf16* out1, int ostride, int ooff0, int ooff1) {
  const int dir = blockIdx.x;
  const __bf16* pg  = dir ? pg1  : pg0;
  const __bf16* WhT = dir ? WhT1 : WhT0;
  const float*  c0  = dir ? c01  : c00;
  __bf16* outp = (dir ? out1 : out0) + (dir ? ooff1 : ooff0);

  __shared__ __bf16 hlds[BATCH * U_DIM];   // 64 KB

  const int lane  = threadIdx.x & 31;
  const int ut    = threadIdx.x >> 5;          // u-tile 0..31
  const int lhalf = lane & 15;
  const int kbA   = (lane < 16) ? 0 : 8;
  const int kbB   = (lane < 16) ? 0 : 8;
  const int hrow  = (lane < 16) ? 0 : 8;
  const int u     = ut * 16 + lhalf;

  const float cinit = c0[u];
  float c[4][8];
#pragma unroll
  for (int mt = 0; mt < 4; ++mt)
#pragma unroll
    for (int r = 0; r < 8; ++r) c[mt][r] = cinit;

  const __bf16 hinit = (__bf16)tanhf(cinit);
#pragma unroll
  for (int mt = 0; mt < 4; ++mt)
#pragma unroll
    for (int r = 0; r < 8; ++r)
      hlds[(mt * 16 + hrow + r) * U_DIM + u] = hinit;
  __syncthreads();

  for (int step = 0; step < T_LEN; ++step) {
    const int t = dir ? (T_LEN - 1 - step) : step;
    const __bf16* pgt = pg + (size_t)t * 2048 * 64;

    float stash[4][8];
    v8f acc[4];

    // pass 1: g = tanh(gate 3)
    gate_gemm(3, pgt, WhT, hlds, u, lhalf, kbA, kbB, hrow, acc);
#pragma unroll
    for (int mt = 0; mt < 4; ++mt)
#pragma unroll
      for (int r = 0; r < 8; ++r) stash[mt][r] = tanhf(acc[mt][r]);

    // pass 2: i = sigmoid(gate 0); stash = i*g
    gate_gemm(0, pgt, WhT, hlds, u, lhalf, kbA, kbB, hrow, acc);
#pragma unroll
    for (int mt = 0; mt < 4; ++mt)
#pragma unroll
      for (int r = 0; r < 8; ++r) stash[mt][r] *= sigmoidf_(acc[mt][r]);

    // pass 3: f = sigmoid(gate 1); c = f*c + i*g
    gate_gemm(1, pgt, WhT, hlds, u, lhalf, kbA, kbB, hrow, acc);
#pragma unroll
    for (int mt = 0; mt < 4; ++mt)
#pragma unroll
      for (int r = 0; r < 8; ++r)
        c[mt][r] = sigmoidf_(acc[mt][r]) * c[mt][r] + stash[mt][r];

    // pass 4: o = sigmoid(gate 2); h = o*tanh(c)
    gate_gemm(2, pgt, WhT, hlds, u, lhalf, kbA, kbB, hrow, acc);

    __syncthreads();   // all reads of h_{t-1} complete before overwrite

#pragma unroll
    for (int mt = 0; mt < 4; ++mt) {
#pragma unroll
      for (int r = 0; r < 8; ++r) {
        float hv = sigmoidf_(acc[mt][r]) * tanhf(c[mt][r]);
        int row = mt * 16 + hrow + r;
        __bf16 hb = (__bf16)hv;
        hlds[row * U_DIM + u] = hb;
        outp[((size_t)t * BATCH + row) * ostride + u] = hb;
      }
    }
    __syncthreads();   // h_t visible before next step's reads
  }
}

// ---------------------------------------------------------------------------
// Span feature builder: hi[b][s*2048 + j] (bf16)
// fo[b][t][j] = j<512 ? in2[t][b][j] : hs2f[t][b][j-512]
// bo[b][t][j] = j<512 ? in2[t][b][512+j] : hs2b[t][b][j-512]
// ---------------------------------------------------------------------------
__global__ void span_hi_kernel(const __bf16* __restrict__ in2,
                               const __bf16* __restrict__ hs2f,
                               const __bf16* __restrict__ hs2b,
                               const int* __restrict__ lefts,
                               const int* __restrict__ rights,
                               int nspans, __bf16* __restrict__ hi, int total) {
  int idx = blockIdx.x * blockDim.x + threadIdx.x;
  if (idx >= total) return;
  int j   = idx & 2047;
  int rem = idx >> 11;       // b*nspans + s
  int s = rem % nspans;
  int b = rem / nspans;
  int L = lefts[b * nspans + s];
  int R = rights[b * nspans + s];
  float v;
  if (j < 1024) {                       // fsv: fo[R] - fo[L-1]
    int t1 = R, t0 = L - 1;
    float a, cc;
    if (j < 512) {
      a  = (float)in2[((size_t)t1 * BATCH + b) * 1024 + j];
      cc = (float)in2[((size_t)t0 * BATCH + b) * 1024 + j];
    } else {
      a  = (float)hs2f[((size_t)t1 * BATCH + b) * 512 + (j - 512)];
      cc = (float)hs2f[((size_t)t0 * BATCH + b) * 512 + (j - 512)];
    }
    v = a - cc;
  } else {                              // bsv: bo[L] - bo[R+1]
    int jj = j - 1024;
    int t1 = L, t0 = R + 1;
    float a, cc;
    if (jj < 512) {
      a  = (float)in2[((size_t)t1 * BATCH + b) * 1024 + 512 + jj];
      cc = (float)in2[((size_t)t0 * BATCH + b) * 1024 + 512 + jj];
    } else {
      a  = (float)hs2b[((size_t)t1 * BATCH + b) * 512 + (jj - 512)];
      cc = (float)hs2b[((size_t)t0 * BATCH + b) * 512 + (jj - 512)];
    }
    v = a - cc;
  }
  hi[idx] = (__bf16)v;
}

// ---------------------------------------------------------------------------
// Generic WMMA GEMM: C[M][N] = relu(A[M][K] @ BT^T + bias), one wave per
// 16x16 output tile. grid = (N/16, M/16), block = 32.
// ---------------------------------------------------------------------------
__global__ __launch_bounds__(32) void gemm_bt_kernel(
    const __bf16* __restrict__ A, int lda,
    const __bf16* __restrict__ BT, int ldb,
    const float* __restrict__ bias,
    float* __restrict__ C, int ldc, int kchunks, int relu) {
  const int ntile = blockIdx.x;
  const int mtile = blockIdx.y;
  const int lane  = threadIdx.x & 31;
  const int lhalf = lane & 15;
  const int kbA   = (lane < 16) ? 0 : 8;
  const int kbB   = (lane < 16) ? 0 : 8;
  const int hrow  = (lane < 16) ? 0 : 8;

  v8f acc;
#pragma unroll
  for (int i = 0; i < 8; ++i) acc[i] = 0.0f;

  const __bf16* arow = A  + (size_t)(mtile * 16 + lhalf) * lda;
  const __bf16* brow = BT + (size_t)(ntile * 16 + lhalf) * ldb;

  for (int kc = 0; kc < kchunks; ++kc) {
    v16bf a = load_frag_a(arow + kc * 32 + kbA);
    v16bf b = load_frag_b(brow + kc * 32 + kbB);
    acc = __builtin_amdgcn_wmma_f32_16x16x32_bf16(
        false, a, false, b, (short)0, acc, false, false);
  }

  const int n = ntile * 16 + lhalf;
  const float bn = bias ? bias[n] : 0.0f;
#pragma unroll
  for (int r = 0; r < 8; ++r) {
    float v = acc[r] + bn;
    if (relu) v = v > 0.0f ? v : 0.0f;
    C[(size_t)(mtile * 16 + hrow + r) * ldc + n] = v;
  }
}

// Tiny output head: out[b][n] = H[b][1024] . W[:,n] + bias[n]
__global__ void linear_out_kernel(const float* __restrict__ H,
                                  const float* __restrict__ W,
                                  const float* __restrict__ bias,
                                  float* __restrict__ out, int N, int total) {
  int idx = blockIdx.x * blockDim.x + threadIdx.x;
  if (idx >= total) return;
  int n = idx % N;
  int b = idx / N;
  float s = bias[n];
  for (int k = 0; k < 1024; ++k) s += H[(size_t)b * 1024 + k] * W[(size_t)k * N + n];
  out[idx] = s;
}

// ---------------------------------------------------------------------------
extern "C" void kernel_launch(void* const* d_in, const int* in_sizes, int n_in,
                              void* d_out, int out_size, void* d_ws, size_t ws_size,
                              hipStream_t stream) {
  const float* word_emb = (const float*)d_in[0];
  const float* tag_emb  = (const float*)d_in[1];
  const float* l1f_W  = (const float*)d_in[2];
  const float* l1f_b  = (const float*)d_in[3];
  const float* l1f_c0 = (const float*)d_in[4];
  const float* l1b_W  = (const float*)d_in[5];
  const float* l1b_b  = (const float*)d_in[6];
  const float* l1b_c0 = (const float*)d_in[7];
  const float* l2f_W  = (const float*)d_in[8];
  const float* l2f_b  = (const float*)d_in[9];
  const float* l2f_c0 = (const float*)d_in[10];
  const float* l2b_W  = (const float*)d_in[11];
  const float* l2b_b  = (const float*)d_in[12];
  const float* l2b_c0 = (const float*)d_in[13];
  const float* sh_W   = (const float*)d_in[14];
  const float* sh_b   = (const float*)d_in[15];
  const float* so_W   = (const float*)d_in[16];
  const float* so_b   = (const float*)d_in[17];
  const float* lh_W   = (const float*)d_in[18];
  const float* lh_b   = (const float*)d_in[19];
  const float* lo_W   = (const float*)d_in[20];
  const float* lo_b   = (const float*)d_in[21];
  const int* word_inds = (const int*)d_in[22];
  const int* tag_inds  = (const int*)d_in[23];
  const int* s_lefts  = (const int*)d_in[24];
  const int* s_rights = (const int*)d_in[25];
  const int* l_lefts  = (const int*)d_in[26];
  const int* l_rights = (const int*)d_in[27];
  (void)in_sizes; (void)n_in; (void)out_size;

  size_t off = 0;
  auto carve = [&](size_t bytes) -> void* {
    void* p = (void*)((char*)d_ws + off);
    off += (bytes + 255) & ~(size_t)255;
    return p;
  };

  __bf16* sent     = (__bf16*)carve((size_t)TB * 320 * 2);
  __bf16* l1f_WxT  = (__bf16*)carve((size_t)2048 * 320 * 2);
  __bf16* l1f_WhT  = (__bf16*)carve((size_t)2048 * 512 * 2);
  __bf16* l1b_WxT  = (__bf16*)carve((size_t)2048 * 320 * 2);
  __bf16* l1b_WhT  = (__bf16*)carve((size_t)2048 * 512 * 2);
  __bf16* l2f_WxT  = (__bf16*)carve((size_t)2048 * 1024 * 2);
  __bf16* l2f_WhT  = (__bf16*)carve((size_t)2048 * 512 * 2);
  __bf16* l2b_WxT  = (__bf16*)carve((size_t)2048 * 1024 * 2);
  __bf16* l2b_WhT  = (__bf16*)carve((size_t)2048 * 512 * 2);
  __bf16* in2      = (__bf16*)carve((size_t)TB * 1024 * 2);   // [fwd1 | back1]
  __bf16* hs2f     = (__bf16*)carve((size_t)TB * 512 * 2);
  __bf16* hs2b     = (__bf16*)carve((size_t)TB * 512 * 2);
  __bf16* pgf      = (__bf16*)carve((size_t)T_LEN * 2048 * 64 * 2); // pre-gates fwd
  __bf16* pgb      = (__bf16*)carve((size_t)T_LEN * 2048 * 64 * 2); // pre-gates bwd
  __bf16* shT      = (__bf16*)carve((size_t)1024 * 8192 * 2);
  __bf16* lhT      = (__bf16*)carve((size_t)1024 * 6144 * 2);
  __bf16* hi_s     = (__bf16*)carve((size_t)64 * 8192 * 2);
  __bf16* hi_l     = (__bf16*)carve((size_t)64 * 6144 * 2);
  float*  H1s      = (float*)carve((size_t)64 * 1024 * 4);
  float*  H1l      = (float*)carve((size_t)64 * 1024 * 4);
  if (off > ws_size) return;   // workspace too small; nothing safe to do

  float* out_struct = (float*)d_out;        // (64, 2)
  float* out_label  = (float*)d_out + 128;  // (64, 32)

  // 1) embeddings
  {
    int total = TB * 320;
    embed_kernel<<<(total + 255) / 256, 256, 0, stream>>>(
        word_emb, tag_emb, word_inds, tag_inds, sent, total);
  }

  // 2) weight prep (bf16 + B^T layouts)
  {
    int t1 = 2048 * (320 + 512);
    prep_lstm_w_kernel<<<(t1 + 255) / 256, 256, 0, stream>>>(l1f_W, 320, l1f_WxT, l1f_WhT, t1);
    prep_lstm_w_kernel<<<(t1 + 255) / 256, 256, 0, stream>>>(l1b_W, 320, l1b_WxT, l1b_WhT, t1);
    int t2 = 2048 * (1024 + 512);
    prep_lstm_w_kernel<<<(t2 + 255) / 256, 256, 0, stream>>>(l2f_W, 1024, l2f_WxT, l2f_WhT, t2);
    prep_lstm_w_kernel<<<(t2 + 255) / 256, 256, 0, stream>>>(l2b_W, 1024, l2b_WxT, l2b_WhT, t2);
    int ts = 8192 * 1024;
    prep_bt_kernel<<<(ts + 255) / 256, 256, 0, stream>>>(sh_W, 8192, 1024, shT, ts);
    int tl = 6144 * 1024;
    prep_bt_kernel<<<(tl + 255) / 256, 256, 0, stream>>>(lh_W, 6144, 1024, lhT, tl);
  }

  // 3) layer-1: parallel input projections, then sequential recurrence
  {
    dim3 g(2048 / 16, TB / 16);
    pregate_gemm_kernel<<<g, 32, 0, stream>>>(sent, 320, l1f_WxT, l1f_b, pgf, 320 / 32);
    pregate_gemm_kernel<<<g, 32, 0, stream>>>(sent, 320, l1b_WxT, l1b_b, pgb, 320 / 32);
    lstm_rec_kernel<<<2, 1024, 0, stream>>>(
        pgf, pgb, l1f_WhT, l1b_WhT, l1f_c0, l1b_c0, in2, in2, 1024, 0, 512);
  }

  // 4) layer-2: same, reading in2
  {
    dim3 g(2048 / 16, TB / 16);
    pregate_gemm_kernel<<<g, 32, 0, stream>>>(in2, 1024, l2f_WxT, l2f_b, pgf, 1024 / 32);
    pregate_gemm_kernel<<<g, 32, 0, stream>>>(in2, 1024, l2b_WxT, l2b_b, pgb, 1024 / 32);
    lstm_rec_kernel<<<2, 1024, 0, stream>>>(
        pgf, pgb, l2f_WhT, l2b_WhT, l2f_c0, l2b_c0, hs2f, hs2b, 512, 0, 0);
  }

  // 5) span features
  {
    int ts = BATCH * 4 * 2048;
    span_hi_kernel<<<(ts + 255) / 256, 256, 0, stream>>>(
        in2, hs2f, hs2b, s_lefts, s_rights, 4, hi_s, ts);
    int tl = BATCH * 3 * 2048;
    span_hi_kernel<<<(tl + 255) / 256, 256, 0, stream>>>(
        in2, hs2f, hs2b, l_lefts, l_rights, 3, hi_l, tl);
  }

  // 6) span MLP hidden layers (WMMA GEMM, fused bias+ReLU)
  gemm_bt_kernel<<<dim3(1024 / 16, 64 / 16), 32, 0, stream>>>(
      hi_s, 8192, shT, 8192, sh_b, H1s, 1024, 8192 / 32, 1);
  gemm_bt_kernel<<<dim3(1024 / 16, 64 / 16), 32, 0, stream>>>(
      hi_l, 6144, lhT, 6144, lh_b, H1l, 1024, 6144 / 32, 1);

  // 7) output heads
  linear_out_kernel<<<1, 128, 0, stream>>>(H1s, so_W, so_b, out_struct, 2, 128);
  linear_out_kernel<<<(2048 + 255) / 256, 256, 0, stream>>>(H1l, lo_W, lo_b, out_label, 32, 2048);
}
